// PRoPEAttention_75204877353703
// MI455X (gfx1250) — matrix operand order repaired and checked by reading, not compiled, not run on hardware
//
#include <hip/hip_runtime.h>
#include <hip/hip_bf16.h>
#include <math.h>

typedef __attribute__((ext_vector_type(16))) __bf16 v16bf;
typedef __attribute__((ext_vector_type(8)))  __bf16 v8bf;
typedef __attribute__((ext_vector_type(8)))  float  v8f;

#define N_TOK 2048
#define DIMC  1024
#define HEADS 16
#define HD    64
#define NCAM  2
#define NPER  1024
#define PW    32

// ---------------------------------------------------------------------------
// CDNA5 async global->LDS copy (ASYNCcnt-tracked), 16B per lane.
// LDS address comes from the VDST operand; generic shared-ptr low 32 bits are the LDS
// byte offset (flat aperture mapping truncates to addr[31:0]).
// ---------------------------------------------------------------------------
__device__ __forceinline__ void async_copy16(unsigned lds_off, const void* gptr) {
  asm volatile("global_load_async_to_lds_b128 %0, %1, off"
               :: "v"(lds_off), "v"((unsigned long long)(uintptr_t)gptr)
               : "memory");
}
__device__ __forceinline__ void wait_async() {
  asm volatile("s_wait_asynccnt 0x0" ::: "memory");
}

// ---------------------------------------------------------------------------
// f32 -> bf16 conversion
// ---------------------------------------------------------------------------
__global__ void cvt_f32_bf16(const float* __restrict__ in, __bf16* __restrict__ out, int n) {
  int i = blockIdx.x * blockDim.x + threadIdx.x;
  if (i < n) out[i] = (__bf16)in[i];
}

// ---------------------------------------------------------------------------
// Per-camera P = Kn @ ext and P^-1 (Gauss-Jordan).  Pm: [cam][0..15]=P,
// [cam][16..31]=Pinv.
// ---------------------------------------------------------------------------
__global__ void compute_P(const float* __restrict__ ext, const float* __restrict__ Ks,
                          float* __restrict__ Pm) {
  int c = threadIdx.x;
  if (c >= NCAM) return;
  const float sx = 2.0f / 448.0f, sy = 2.0f / 448.0f;
  const float* K3 = Ks + c * 9;
  const float* E  = ext + c * 16;
  float Kn[16];
  for (int i = 0; i < 16; ++i) Kn[i] = 0.0f;
  Kn[0]  = K3[0] * sx;
  Kn[2]  = K3[2] * sx - 1.0f;
  Kn[5]  = K3[4] * sy;
  Kn[6]  = K3[5] * sy - 1.0f;
  Kn[10] = 1.0f; Kn[15] = 1.0f;
  float P[16];
  for (int i = 0; i < 4; ++i)
    for (int j = 0; j < 4; ++j) {
      float s = 0.0f;
      for (int k = 0; k < 4; ++k) s += Kn[i*4+k] * E[k*4+j];
      P[i*4+j] = s;
    }
  float A[16], I[16];
  for (int i = 0; i < 16; ++i) { A[i] = P[i]; I[i] = 0.0f; }
  I[0] = I[5] = I[10] = I[15] = 1.0f;
  for (int col = 0; col < 4; ++col) {
    int piv = col; float best = fabsf(A[col*4+col]);
    for (int r = col + 1; r < 4; ++r) {
      float v = fabsf(A[r*4+col]);
      if (v > best) { best = v; piv = r; }
    }
    if (piv != col)
      for (int j = 0; j < 4; ++j) {
        float t = A[col*4+j]; A[col*4+j] = A[piv*4+j]; A[piv*4+j] = t;
        t = I[col*4+j]; I[col*4+j] = I[piv*4+j]; I[piv*4+j] = t;
      }
    float d = 1.0f / A[col*4+col];
    for (int j = 0; j < 4; ++j) { A[col*4+j] *= d; I[col*4+j] *= d; }
    for (int r = 0; r < 4; ++r) if (r != col) {
      float f = A[r*4+col];
      for (int j = 0; j < 4; ++j) { A[r*4+j] -= f * A[col*4+j]; I[r*4+j] -= f * I[col*4+j]; }
    }
  }
  for (int i = 0; i < 16; ++i) { Pm[c*32 + i] = P[i]; Pm[c*32 + 16 + i] = I[i]; }
}

// ---------------------------------------------------------------------------
// A 16x32 bf16 fragment (ISA 7.12.2): lane L holds row (L&15);
// elem e -> K = 16*(L>=16)+e (contiguous, matching B fragments).
// ---------------------------------------------------------------------------
__device__ __forceinline__ v16bf load_a16(const __bf16* rowk, int half) {
  v8bf lo = *(const v8bf*)(rowk + half * 16);
  v8bf hi = *(const v8bf*)(rowk + half * 16 + 8);
  return __builtin_shufflevector(lo, hi, 0,1,2,3,4,5,6,7,8,9,10,11,12,13,14,15);
}

// ---------------------------------------------------------------------------
// C[M,N] = A[M,K] @ Bw[N,K]^T (+bias).  One wave per 32x32 output tile:
// 2 A-frags x 2 B-frags -> 4 independent WMMA accumulators per K-step
// (2x the FLOP/byte of a 16x16 tile; 4-deep WMMA pipelining).
// B 32x16 bf16: lane L holds col (L&15); elem e -> K = 16*(L>=16)+e.
// ---------------------------------------------------------------------------
__global__ __launch_bounds__(32)
void wmma_gemm_nt(const __bf16* __restrict__ A, const __bf16* __restrict__ Bw,
                  const float* __restrict__ bias, float* __restrict__ C,
                  int M, int N, int K) {
  int ntN  = N >> 5;
  int m0   = (blockIdx.x / ntN) << 5;
  int n0   = (blockIdx.x % ntN) << 5;
  int lane = threadIdx.x;
  int half = lane >> 4, l15 = lane & 15;
  const __bf16* arow0 = A  + (size_t)(m0 + l15) * K;
  const __bf16* arow1 = A  + (size_t)(m0 + 16 + l15) * K;
  const __bf16* brow0 = Bw + (size_t)(n0 + l15) * K;
  const __bf16* brow1 = Bw + (size_t)(n0 + 16 + l15) * K;
  v8f acc00 = {}, acc01 = {}, acc10 = {}, acc11 = {};
  for (int k0 = 0; k0 < K; k0 += 32) {
    __builtin_prefetch(arow0 + k0 + 256, 0, 1);
    __builtin_prefetch(arow1 + k0 + 256, 0, 1);
    __builtin_prefetch(brow0 + k0 + 256, 0, 1);
    __builtin_prefetch(brow1 + k0 + 256, 0, 1);
    v16bf a0 = load_a16(arow0 + k0, half);
    v16bf a1 = load_a16(arow1 + k0, half);
    v16bf b0 = *(const v16bf*)(brow0 + k0 + half * 16);
    v16bf b1 = *(const v16bf*)(brow1 + k0 + half * 16);
    acc00 = __builtin_amdgcn_wmma_f32_16x16x32_bf16(false, a0, false, b0, (short)0, acc00, false, false);
    acc01 = __builtin_amdgcn_wmma_f32_16x16x32_bf16(false, a0, false, b1, (short)0, acc01, false, false);
    acc10 = __builtin_amdgcn_wmma_f32_16x16x32_bf16(false, a1, false, b0, (short)0, acc10, false, false);
    acc11 = __builtin_amdgcn_wmma_f32_16x16x32_bf16(false, a1, false, b1, (short)0, acc11, false, false);
  }
  int cn0 = n0 + l15, cn1 = n0 + 16 + l15;
  float bv0 = bias ? bias[cn0] : 0.0f;
  float bv1 = bias ? bias[cn1] : 0.0f;
#pragma unroll
  for (int r = 0; r < 8; ++r) {
    size_t row0 = (size_t)(m0 + r + half * 8) * N;
    size_t row1 = (size_t)(m0 + 16 + r + half * 8) * N;
    C[row0 + cn0] = acc00[r] + bv0;
    C[row0 + cn1] = acc01[r] + bv1;
    C[row1 + cn0] = acc10[r] + bv0;
    C[row1 + cn1] = acc11[r] + bv1;
  }
}

// ---------------------------------------------------------------------------
// PRoPE transform of q/k/v.  qkv f32 [n][s*1024 + h*64 + d].
// q -> Qh[h][n][d] (bf16, pre-scaled 1/sqrt(64)); k -> Kh[h][n][d];
// v -> Vt[h][d][n] (transposed for contiguous PV B-fragments).
// ---------------------------------------------------------------------------
__global__ void prope_apply(const float* __restrict__ qkv, const float* __restrict__ Pm,
                            __bf16* __restrict__ Qh, __bf16* __restrict__ Kh,
                            __bf16* __restrict__ Vt) {
  int idx = blockIdx.x * blockDim.x + threadIdx.x;
  if (idx >= N_TOK * HEADS) return;
  int n = idx >> 4;
  int h = idx & 15;
  int cam = n / NPER;
  int t   = n % NPER;
  float px = (float)(t % PW), py = (float)(t / PW);
  const float* P  = Pm + cam * 32;
  const float* Pi = P + 16;
  float cs[16], sn[16];
#pragma unroll
  for (int f = 0; f < 8; ++f) {
    float fr = __powf(100.0f, -(float)f / 8.0f);
    float ax = px * fr, ay = py * fr;
    cs[f]     = __cosf(ax); sn[f]     = __sinf(ax);
    cs[8 + f] = __cosf(ay); sn[8 + f] = __sinf(ay);
  }
  const float* base = qkv + (size_t)n * (3 * DIMC);
  for (int s = 0; s < 3; ++s) {
    const float* tv = base + s * DIMC + h * HD;
    float o[64];
#pragma unroll
    for (int g = 0; g < 8; ++g) {
      float t0 = tv[g*4], t1 = tv[g*4+1], t2 = tv[g*4+2], t3 = tv[g*4+3];
#pragma unroll
      for (int i = 0; i < 4; ++i) {
        float m0, m1, m2, m3;
        if (s == 0) { m0 = Pi[0*4+i]; m1 = Pi[1*4+i]; m2 = Pi[2*4+i]; m3 = Pi[3*4+i]; } // Mq=Pinv^T
        else        { m0 = P[i*4+0];  m1 = P[i*4+1];  m2 = P[i*4+2];  m3 = P[i*4+3];  } // Mk=P
        o[g*4+i] = m0*t0 + m1*t1 + m2*t2 + m3*t3;
      }
    }
#pragma unroll
    for (int p = 0; p < 16; ++p) {
      float a = tv[32 + 2*p], b = tv[32 + 2*p + 1];
      o[32 + 2*p]     = a * cs[p] - b * sn[p];
      o[32 + 2*p + 1] = a * sn[p] + b * cs[p];
    }
    if (s == 0) {
      __bf16* dst = Qh + ((size_t)h * N_TOK + n) * HD;
      for (int d = 0; d < HD; ++d) dst[d] = (__bf16)(o[d] * 0.125f);
    } else if (s == 1) {
      __bf16* dst = Kh + ((size_t)h * N_TOK + n) * HD;
      for (int d = 0; d < HD; ++d) dst[d] = (__bf16)o[d];
    } else {
      for (int d = 0; d < HD; ++d)
        Vt[((size_t)h * HD + d) * N_TOK + n] = (__bf16)o[d];
    }
  }
}

// ---------------------------------------------------------------------------
// Flash attention, 4 waves / block, 64 Q-rows per (head, block).
// K/V 32-key blocks staged in LDS with async global->LDS copies (double
// buffered, ASYNCcnt + barrier), shared by all 4 waves (4x L2 traffic cut).
// Per wave per block: 4 WMMAs for S, online softmax with 16-lane shfl_xor row
// reductions, P relayout C/D->A via per-wave LDS tile, 4 WMMAs for O += P@V.
// ---------------------------------------------------------------------------
__global__ __launch_bounds__(128)
void flash_attn(const __bf16* __restrict__ Qh, const __bf16* __restrict__ Kh,
                const __bf16* __restrict__ Vt, float* __restrict__ Oout) {
  __shared__ __align__(128) __bf16 kbuf[2][32 * 64];   // [key][d]   4KB each
  __shared__ __align__(128) __bf16 vbuf[2][64 * 32];   // [d][key]   4KB each
  __shared__ __align__(128) __bf16 pt[4][16 * 32];     // per-wave P tile
  int h    = blockIdx.y;
  int m0   = blockIdx.x << 6;            // 64 rows / block
  int tid  = threadIdx.x;
  int wave = tid >> 5;
  int lane = tid & 31;
  int half = lane >> 4, l15 = lane & 15;

  const __bf16* Kbase = Kh + (size_t)h * N_TOK * HD;
  const char*   Vbase = (const char*)(Vt + (size_t)h * HD * N_TOK);

  const __bf16* qrow = Qh + ((size_t)h * N_TOK + m0 + wave * 16 + l15) * HD;
  v16bf aq0 = load_a16(qrow, half);
  v16bf aq1 = load_a16(qrow + 32, half);

  v8f o0 = {}, o1 = {}, o2 = {}, o3 = {};
  float mrow[8], lrow[8];
#pragma unroll
  for (int r = 0; r < 8; ++r) { mrow[r] = -1e30f; lrow[r] = 0.0f; }

  // Stage one 32-key block: K 4KB + V 4KB = 512 x 16B transfers,
  // 4 async instructions per wave (each wave covers its quarter).
  auto issue = [&](int buf, int j) {
    int id0 = wave * 64 + lane;
    const char* kg = (const char*)(Kbase + (size_t)j * HD);
    unsigned kl = (unsigned)(uintptr_t)&kbuf[buf][0];
    async_copy16(kl + id0 * 16,        kg + id0 * 16);
    async_copy16(kl + (id0 + 32) * 16, kg + (id0 + 32) * 16);
    const char* vg = Vbase + (size_t)j * 2;
    unsigned vl = (unsigned)(uintptr_t)&vbuf[buf][0];
    int d0 = id0 >> 2, s0 = id0 & 3;
    int d1 = (id0 + 32) >> 2, s1 = (id0 + 32) & 3;
    async_copy16(vl + d0 * 64 + s0 * 16, vg + (size_t)d0 * 4096 + s0 * 16);
    async_copy16(vl + d1 * 64 + s1 * 16, vg + (size_t)d1 * 4096 + s1 * 16);
  };

  issue(0, 0);
  int buf = 0;
  for (int j = 0; j < N_TOK; j += 32, buf ^= 1) {
    wait_async();
    __syncthreads();
    if (j + 32 < N_TOK) issue(buf ^ 1, j + 32);

    const __bf16* kb = &kbuf[buf][0];
    v16bf bklo0 = *(const v16bf*)(kb + l15 * HD + half * 16);
    v16bf bklo1 = *(const v16bf*)(kb + l15 * HD + 32 + half * 16);
    v16bf bkhi0 = *(const v16bf*)(kb + (16 + l15) * HD + half * 16);
    v16bf bkhi1 = *(const v16bf*)(kb + (16 + l15) * HD + 32 + half * 16);
    v8f slo = {}, shi = {};
    slo = __builtin_amdgcn_wmma_f32_16x16x32_bf16(false, aq0, false, bklo0, (short)0, slo, false, false);
    slo = __builtin_amdgcn_wmma_f32_16x16x32_bf16(false, aq1, false, bklo1, (short)0, slo, false, false);
    shi = __builtin_amdgcn_wmma_f32_16x16x32_bf16(false, aq0, false, bkhi0, (short)0, shi, false, false);
    shi = __builtin_amdgcn_wmma_f32_16x16x32_bf16(false, aq1, false, bkhi1, (short)0, shi, false, false);
#pragma unroll
    for (int r = 0; r < 8; ++r) {
      float sl = slo[r], sh = shi[r];
      float mx = fmaxf(sl, sh);
#pragma unroll
      for (int off = 8; off >= 1; off >>= 1)
        mx = fmaxf(mx, __shfl_xor(mx, off, 32));
      float mnew  = fmaxf(mrow[r], mx);
      float pl    = __expf(sl - mnew), ph = __expf(sh - mnew);
      float alpha = __expf(mrow[r] - mnew);
      mrow[r] = mnew;
      float rs = pl + ph;
#pragma unroll
      for (int off = 8; off >= 1; off >>= 1)
        rs += __shfl_xor(rs, off, 32);
      lrow[r] = lrow[r] * alpha + rs;
      o0[r] *= alpha; o1[r] *= alpha; o2[r] *= alpha; o3[r] *= alpha;
      int row = r + half * 8;
      pt[wave][row * 32 + l15]      = (__bf16)pl;
      pt[wave][row * 32 + 16 + l15] = (__bf16)ph;
    }
    // per-wave LDS tile: DS ops are in-order within a wave; compiler inserts
    // the dscnt wait for the aliasing reload below.
    v16bf ap = load_a16(&pt[wave][0] + l15 * 32, half);
    const __bf16* vb = &vbuf[buf][0];
    v16bf bv0 = *(const v16bf*)(vb + (0 * 16 + l15) * 32 + half * 16);
    v16bf bv1 = *(const v16bf*)(vb + (1 * 16 + l15) * 32 + half * 16);
    v16bf bv2 = *(const v16bf*)(vb + (2 * 16 + l15) * 32 + half * 16);
    v16bf bv3 = *(const v16bf*)(vb + (3 * 16 + l15) * 32 + half * 16);
    o0 = __builtin_amdgcn_wmma_f32_16x16x32_bf16(false, ap, false, bv0, (short)0, o0, false, false);
    o1 = __builtin_amdgcn_wmma_f32_16x16x32_bf16(false, ap, false, bv1, (short)0, o1, false, false);
    o2 = __builtin_amdgcn_wmma_f32_16x16x32_bf16(false, ap, false, bv2, (short)0, o2, false, false);
    o3 = __builtin_amdgcn_wmma_f32_16x16x32_bf16(false, ap, false, bv3, (short)0, o3, false, false);
    __syncthreads();   // all waves done reading this buffer before re-fill
  }
#pragma unroll
  for (int r = 0; r < 8; ++r) {
    float inv = 1.0f / lrow[r];
    int row = m0 + wave * 16 + r + half * 8;
    float* od = Oout + (size_t)row * DIMC + h * HD + l15;
    od[0]  = o0[r] * inv;
    od[16] = o1[r] * inv;
    od[32] = o2[r] * inv;
    od[48] = o3[r] * inv;
  }
}

// ---------------------------------------------------------------------------
// Inverse PRoPE (Mo = Pinv, sign = -1); f32 -> bf16.
// ---------------------------------------------------------------------------
__global__ void out_transform(const float* __restrict__ O, const float* __restrict__ Pm,
                              __bf16* __restrict__ Ab) {
  int idx = blockIdx.x * blockDim.x + threadIdx.x;
  if (idx >= N_TOK * HEADS) return;
  int n = idx >> 4;
  int h = idx & 15;
  int cam = n / NPER;
  int t   = n % NPER;
  float px = (float)(t % PW), py = (float)(t / PW);
  const float* Pi = Pm + cam * 32 + 16;
  float cs[16], sn[16];
#pragma unroll
  for (int f = 0; f < 8; ++f) {
    float fr = __powf(100.0f, -(float)f / 8.0f);
    float ax = px * fr, ay = py * fr;
    cs[f]     = __cosf(ax); sn[f]     = __sinf(ax);
    cs[8 + f] = __cosf(ay); sn[8 + f] = __sinf(ay);
  }
  const float* tv = O + (size_t)n * DIMC + h * HD;
  __bf16* dst = Ab + (size_t)n * DIMC + h * HD;
#pragma unroll
  for (int g = 0; g < 8; ++g) {
    float t0 = tv[g*4], t1 = tv[g*4+1], t2 = tv[g*4+2], t3 = tv[g*4+3];
#pragma unroll
    for (int i = 0; i < 4; ++i)
      dst[g*4+i] = (__bf16)(Pi[i*4+0]*t0 + Pi[i*4+1]*t1 + Pi[i*4+2]*t2 + Pi[i*4+3]*t3);
  }
#pragma unroll
  for (int p = 0; p < 16; ++p) {
    float a = tv[32 + 2*p], b = tv[32 + 2*p + 1];
    dst[32 + 2*p]     = (__bf16)( a * cs[p] + b * sn[p]);
    dst[32 + 2*p + 1] = (__bf16)(-a * sn[p] + b * cs[p]);
  }
}

// ---------------------------------------------------------------------------
extern "C" void kernel_launch(void* const* d_in, const int* in_sizes, int n_in,
                              void* d_out, int out_size, void* d_ws, size_t ws_size,
                              hipStream_t stream) {
  const float* x     = (const float*)d_in[0];
  const float* ext   = (const float*)d_in[1];
  const float* Ks    = (const float*)d_in[2];
  const float* qkvw  = (const float*)d_in[3];
  const float* projw = (const float*)d_in[4];
  const float* projb = (const float*)d_in[5];

  char* ws = (char*)d_ws;
  size_t off = 0;
  auto alloc = [&](size_t bytes) -> void* {
    void* p = ws + off;
    off += (bytes + 255) & ~(size_t)255;
    return p;
  };
  __bf16* xb     = (__bf16*)alloc((size_t)N_TOK * DIMC * 2);
  __bf16* wqkvb  = (__bf16*)alloc((size_t)3 * DIMC * DIMC * 2);
  __bf16* wprojb = (__bf16*)alloc((size_t)DIMC * DIMC * 2);
  float*  Pm     = (float*) alloc((size_t)NCAM * 32 * 4);
  float*  qkv    = (float*) alloc((size_t)N_TOK * 3 * DIMC * 4);
  __bf16* Qh     = (__bf16*)alloc((size_t)HEADS * N_TOK * HD * 2);
  __bf16* Kh     = (__bf16*)alloc((size_t)HEADS * N_TOK * HD * 2);
  __bf16* Vt     = (__bf16*)alloc((size_t)HEADS * HD * N_TOK * 2);
  float*  Oout   = (float*) alloc((size_t)N_TOK * DIMC * 4);
  __bf16* Ab     = (__bf16*)alloc((size_t)N_TOK * DIMC * 2);

  cvt_f32_bf16<<<(N_TOK * DIMC + 255) / 256, 256, 0, stream>>>(x, xb, N_TOK * DIMC);
  cvt_f32_bf16<<<(3 * DIMC * DIMC + 255) / 256, 256, 0, stream>>>(qkvw, wqkvb, 3 * DIMC * DIMC);
  cvt_f32_bf16<<<(DIMC * DIMC + 255) / 256, 256, 0, stream>>>(projw, wprojb, DIMC * DIMC);
  compute_P<<<1, 32, 0, stream>>>(ext, Ks, Pm);

  // qkv = x @ qkv_w^T : [2048,1024] x [3072,1024]^T, 32x32 tiles per wave
  wmma_gemm_nt<<<(N_TOK / 32) * (3 * DIMC / 32), 32, 0, stream>>>(
      xb, wqkvb, nullptr, qkv, N_TOK, 3 * DIMC, DIMC);

  prope_apply<<<(N_TOK * HEADS + 255) / 256, 256, 0, stream>>>(qkv, Pm, Qh, Kh, Vt);

  dim3 fg(N_TOK / 64, HEADS);
  flash_attn<<<fg, 128, 0, stream>>>(Qh, Kh, Vt, Oout);

  out_transform<<<(N_TOK * HEADS + 255) / 256, 256, 0, stream>>>(Oout, Pm, Ab);

  // out = Ab @ proj_w^T + proj_b
  wmma_gemm_nt<<<(N_TOK / 32) * (DIMC / 32), 32, 0, stream>>>(
      Ab, wprojb, projb, (float*)d_out, N_TOK, DIMC, DIMC);
}
